// BinarizedLSTM_90099823935566
// MI455X (gfx1250) — matrix operand order repaired and checked by reading, not compiled, not run on hardware
//
#include <hip/hip_runtime.h>
#include <hip/hip_bf16.h>

typedef __attribute__((ext_vector_type(16))) _Float16 v16h;
typedef __attribute__((ext_vector_type(8)))  _Float16 v8h;
typedef __attribute__((ext_vector_type(8)))  float    v8f;

#define WMMA_F16(a,b,c) \
  __builtin_amdgcn_wmma_f32_16x16x32_f16(false,(a),false,(b),(short)0,(c),false,false)

constexpr int   Bsz = 64, T = 1024, D = 512, H = 512, G = 2048;
constexpr int   NT  = 128;   // 16-wide gate-column tiles (2048/16)
constexpr int   KT  = 16;    // 32-deep K tiles (512/32)
constexpr int   MT  = 4;     // 16-row batch tiles (64/16)
constexpr float EPS = 1e-5f;

__device__ __forceinline__ float sigf(float x) { return 1.f / (1.f + __expf(-x)); }

// ---------------------------------------------------------------------------
// Binarize weights and pre-pack into the WMMA B-fragment layout (ISA 7.12.2):
// 16-bit B 32x16: lanes 0-15 hold K=0..7,16..23 (N=lane), lanes 16-31 hold K=8..15,24..31.
// Layout: [layer][ntile][ktile][lane][half]  (16 contiguous halves per lane)
// ---------------------------------------------------------------------------
__global__ void pack_weights_kernel(const float* __restrict__ wih,
                                    const float* __restrict__ whh,
                                    _Float16* __restrict__ wihF,
                                    _Float16* __restrict__ whhF) {
  unsigned tid = blockIdx.x * blockDim.x + threadIdx.x;  // 1<<22 threads
  int half = tid & 15;
  int lane = (tid >> 4) & 31;
  int kt   = (tid >> 9) & 15;
  int nt   = (tid >> 13) & 127;
  int mat  = (tid >> 20) & 1;
  int l    = (tid >> 21) & 1;
  int N = lane & 15;
  int K = (half & 7) + 8 * ((lane >> 4) + (half >= 8 ? 2 : 0));  // 0..31 within the K tile
  int k = kt * 32 + K;
  int n = nt * 16 + N;
  const float* src = (mat ? whh : wih) + (((size_t)l * G + n) * D + k);
  float v = *src;
  float s = (v > 0.f) ? 1.f : ((v < 0.f) ? -1.f : 0.f);
  _Float16* dst = mat ? whhF : wihF;
  dst[((((size_t)l * NT + nt) * KT + kt) * 32 + lane) * 16 + half] = (_Float16)s;
}

// ---------------------------------------------------------------------------
// Pre-pack x into WMMA A-fragment layout (16-bit A 16x32, ISA 7.12.2):
// lane<16: halves 0-7 -> K 0-7, 8-15 -> K 16-23 ; lane>=16: K 8-15 / 24-31.
// Layout: [t][mtile][ktile][lane][half]
// ---------------------------------------------------------------------------
__global__ void pack_x_kernel(const float* __restrict__ x,
                              _Float16* __restrict__ xF) {
  unsigned tid = blockIdx.x * blockDim.x + threadIdx.x;  // 1<<25 threads
  int half = tid & 15;
  int lane = (tid >> 4) & 31;
  int kt   = (tid >> 9) & 15;
  int mt   = (tid >> 13) & 3;
  int t    = tid >> 15;
  int M      = lane & 15;
  int hiLane = lane >> 4;
  int K = half + 8 * (hiLane + (half >= 8 ? 1 : 0));
  int k = kt * 32 + K;
  int b = mt * 16 + M;
  xF[tid] = (_Float16)x[((size_t)b * T + t) * D + k];
}

// ---------------------------------------------------------------------------
// Input-projection GEMM: gi = A @ sign(Wih)^T + (b_ih + b_hh), chip-wide.
// A is fragment-packed (xF or layer-1 hF). Fully unrolled K loop: a straight-
// line body lets the scheduler hoist global_load_b128 pairs ahead of their
// WMMAs without loop-carried copies. 8 independent accumulators per wave keep
// the in-place WMMA D-register reuse distance outside the f16 hazard window.
// Output f16 in D-fragment order: [rowtile][ntile][lane][8].
// Grid (T*MT, 2), block 256 (8 waves x 8 ntiles).
// ---------------------------------------------------------------------------
__global__ __launch_bounds__(256)
void gi_gemm_kernel(const _Float16* __restrict__ aF,
                    const _Float16* __restrict__ wF,
                    const float* __restrict__ bih,
                    const float* __restrict__ bhh,
                    _Float16* __restrict__ giF) {
  const int rt   = blockIdx.x;              // row tile = t*4 + mtile
  const int wv   = threadIdx.x >> 5;
  const int lane = threadIdx.x & 31;
  const int nt0  = blockIdx.y * 64 + wv * 8;
  const _Float16* aBase = aF + (size_t)rt * (KT * 32 * 16);

  v8f acc[8] = {};
#pragma unroll
  for (int kk = 0; kk < KT; ++kk) {
    v16h a = *(const v16h*)(aBase + (kk * 32 + lane) * 16);
    if (kk + 1 < KT)  // keep streaming A near (gfx1250 global_prefetch, SE scope)
      __builtin_prefetch(aBase + ((kk + 1) * 32 + lane) * 16, 0, 3);
#pragma unroll
    for (int q = 0; q < 8; ++q) {
      v16h bfr = *(const v16h*)(wF + (((size_t)(nt0 + q) * KT + kk) * 32 + lane) * 16);
      acc[q] = WMMA_F16(a, bfr, acc[q]);
    }
  }
  const int laneLo = lane & 15;
#pragma unroll
  for (int q = 0; q < 8; ++q) {
    int col = (nt0 + q) * 16 + laneLo;
    float bias = bih[col] + bhh[col];
    v8h o;
#pragma unroll
    for (int e = 0; e < 8; ++e) o[e] = (_Float16)(acc[q][e] + bias);
    *(v8h*)(giF + (((size_t)rt * NT + (nt0 + q)) * 32 + lane) * 8) = o;
  }
}

// ---------------------------------------------------------------------------
// Persistent recurrent scan. Grid = 4 (one 16-row batch strip per WG, fully
// independent scans), block = 256 (8 waves). Each wave owns 4 hidden-column
// blocks and their 4 gate tiles -> 16 f32 accumulators. h lives in LDS in
// A-fragment order; Whh fragments stream from L2 (2 MB, resident). The K loop
// is fully unrolled: 256 straight-line WMMAs, loads hoisted by the scheduler,
// 16-accumulator interleave keeps every WMMA outside the hazard window.
// ---------------------------------------------------------------------------
template <int LAYER>
__global__ __launch_bounds__(256)
void lstm_scan_kernel(const _Float16* __restrict__ giF,
                      const _Float16* __restrict__ whhF,
                      const float* __restrict__ lnG,
                      const float* __restrict__ lnB,
                      _Float16* __restrict__ hNextF,   // layer0: packed output for layer1
                      float* __restrict__ out) {
  __shared__ _Float16 hfrag[KT * 32 * 16];  // 16 KB: h_t in A-fragment order
  __shared__ float    redS[16][128];
  __shared__ float    redQ[16][128];
  __shared__ float    meanS[16];
  __shared__ float    rstdS[16];

  const int mtile  = blockIdx.x;
  const int wv     = threadIdx.x >> 5;
  const int lane   = threadIdx.x & 31;
  const int laneLo = lane & 15;
  const int hiL    = lane >> 4;
  const int jbase  = wv * 4;

  for (int i = threadIdx.x; i < KT * 32 * 16; i += 256) hfrag[i] = (_Float16)0.f;

  // Cache LN params + scatter constants (D-layout col -> A-fragment position).
  float gam[4][4], bet[4][4];  // [gate][q]
  int scol[4], skt[4], shalf[4], slAdd[4];
#pragma unroll
  for (int q = 0; q < 4; ++q) {
    int colH = (jbase + q) * 16 + laneLo;
    scol[q] = colH;
    int K = colH & 31, seg = K >> 3;
    skt[q]   = colH >> 5;
    shalf[q] = (K & 7) + ((seg & 2) ? 8 : 0);
    slAdd[q] = (seg & 1) ? 16 : 0;
#pragma unroll
    for (int g = 0; g < 4; ++g) {
      gam[g][q] = lnG[g * H + colH];
      bet[g][q] = lnB[g * H + colH];
    }
  }

  v8f cst[4] = {};  // cell state, per column block
  __syncthreads();

  for (int t = 0; t < T; ++t) {
    // Init accumulators from precomputed input projection (bias folded in).
    v8f acc[4][4];  // [gate][q]
    const _Float16* gbase = giF + (((size_t)t * MT + mtile) * NT) * 32 * 8;
#pragma unroll
    for (int g = 0; g < 4; ++g)
#pragma unroll
      for (int q = 0; q < 4; ++q) {
        int gt = g * 32 + jbase + q;
        v8h gv = *(const v8h*)(gbase + ((size_t)gt * 32 + lane) * 8);
#pragma unroll
        for (int e = 0; e < 8; ++e) acc[g][q][e] = (float)gv[e];
      }

    // Prefetch next timestep's gi block (64 KB/WG) behind this step's WMMAs.
    if (t + 1 < T) {
      const char* nb = (const char*)(giF + (((size_t)(t + 1) * MT + mtile) * NT) * 32 * 8);
      __builtin_prefetch(nb + threadIdx.x * 256, 0, 3);
      __builtin_prefetch(nb + threadIdx.x * 256 + 128, 0, 3);
    }

    // gh = h @ sign(Whh)^T : 16 K-steps x 16 tiles = 256 WMMAs / wave / step.
#pragma unroll
    for (int kk = 0; kk < KT; ++kk) {
      v16h a = *(const v16h*)&hfrag[(kk * 32 + lane) * 16];
#pragma unroll
      for (int g = 0; g < 4; ++g)
#pragma unroll
        for (int q = 0; q < 4; ++q) {
          int gt = g * 32 + jbase + q;
          v16h bfr = *(const v16h*)(whhF + (((size_t)gt * KT + kk) * 32 + lane) * 16);
          acc[g][q] = WMMA_F16(a, bfr, acc[g][q]);
        }
    }

    // LayerNorm over the 2048 gate columns of each of the 16 rows.
#pragma unroll
    for (int e = 0; e < 8; ++e) {
      float sv = 0.f, sq = 0.f;
#pragma unroll
      for (int g = 0; g < 4; ++g)
#pragma unroll
        for (int q = 0; q < 4; ++q) { float v = acc[g][q][e]; sv += v; sq += v * v; }
      int row = e + 8 * hiL;
      redS[row][wv * 16 + laneLo] = sv;
      redQ[row][wv * 16 + laneLo] = sq;
    }
    __syncthreads();
    if (threadIdx.x < 16) {
      int row = threadIdx.x;
      float sv = 0.f, sq = 0.f;
      for (int i = 0; i < 128; ++i) { sv += redS[row][i]; sq += redQ[row][i]; }
      float m   = sv * (1.f / G);
      float var = sq * (1.f / G) - m * m;
      meanS[row] = m;
      rstdS[row] = rsqrtf(var + EPS);
    }
    __syncthreads();

    // Gates + state update; scatter hy back into LDS A-fragment buffer.
#pragma unroll
    for (int q = 0; q < 4; ++q) {
#pragma unroll
      for (int e = 0; e < 8; ++e) {
        int row = e + 8 * hiL;
        float m = meanS[row], rs = rstdS[row];
        float r  = (acc[0][q][e] - m) * rs * gam[0][q] + bet[0][q];
        float ii = (acc[1][q][e] - m) * rs * gam[1][q] + bet[1][q];
        float nn = (acc[2][q][e] - m) * rs * gam[2][q] + bet[2][q];
        float cg = (acc[3][q][e] - m) * rs * gam[3][q] + bet[3][q];
        float cy = sigf(cg) * cst[q][e] + sigf(ii) * tanhf(nn);
        cst[q][e] = cy;
        float hy = sigf(r) * tanhf(cy);
        hfrag[(skt[q] * 32 + row + slAdd[q]) * 16 + shalf[q]] = (_Float16)hy;
        if (LAYER == 1) {
          int b = mtile * 16 + row;
          out[((size_t)b * T + t) * H + scol[q]] = hy;
        }
        if (t == T - 1) {
          int b = mtile * 16 + row;
          float* hs = out + (size_t)Bsz * T * H;
          float* cs = hs + (size_t)2 * Bsz * H;
          hs[((size_t)LAYER * Bsz + b) * H + scol[q]] = hy;
          cs[((size_t)LAYER * Bsz + b) * H + scol[q]] = cy;
        }
      }
    }
    __syncthreads();

    if (LAYER == 0) {  // export packed h_t for layer-1's projection GEMM
      const uint4* s = (const uint4*)hfrag;
      uint4* d = (uint4*)(hNextF + ((size_t)t * MT + mtile) * (KT * 32 * 16));
      for (int i = threadIdx.x; i < (KT * 32 * 16) / 8; i += 256) d[i] = s[i];
    }
  }
}

// ---------------------------------------------------------------------------
extern "C" void kernel_launch(void* const* d_in, const int* in_sizes, int n_in,
                              void* d_out, int out_size, void* d_ws, size_t ws_size,
                              hipStream_t stream) {
  (void)in_sizes; (void)n_in; (void)out_size; (void)ws_size;
  const float* x    = (const float*)d_in[0];
  const float* w_ih = (const float*)d_in[1];
  const float* w_hh = (const float*)d_in[2];
  const float* b_ih = (const float*)d_in[3];
  const float* b_hh = (const float*)d_in[4];
  const float* ln_g = (const float*)d_in[5];
  const float* ln_b = (const float*)d_in[6];
  float* out = (float*)d_out;

  _Float16* ws = (_Float16*)d_ws;
  const size_t WFRAG = (size_t)NT * KT * 32 * 16;       // 1 Mi halves / matrix / layer
  const size_t AFRAG = (size_t)T * MT * KT * 32 * 16;   // 32 Mi halves
  _Float16* wihF = ws;                  //   4 MB
  _Float16* whhF = wihF + 2 * WFRAG;    //   4 MB
  _Float16* xF   = whhF + 2 * WFRAG;    //  64 MB
  _Float16* h1F  = xF + AFRAG;          //  64 MB
  _Float16* giF  = h1F + AFRAG;         // 256 MB (reused across layers)

  pack_weights_kernel<<<(1u << 22) / 256, 256, 0, stream>>>(w_ih, w_hh, wihF, whhF);
  pack_x_kernel<<<(1u << 25) / 256, 256, 0, stream>>>(x, xF);

  // Layer 0
  gi_gemm_kernel<<<dim3(T * MT, 2), 256, 0, stream>>>(xF, wihF, b_ih, b_hh, giF);
  lstm_scan_kernel<0><<<MT, 256, 0, stream>>>(giF, whhF, ln_g, ln_b, h1F, out);

  // Layer 1
  gi_gemm_kernel<<<dim3(T * MT, 2), 256, 0, stream>>>(h1F, wihF + WFRAG,
                                                      b_ih + G, b_hh + G, giF);
  lstm_scan_kernel<1><<<MT, 256, 0, stream>>>(giF, whhF + WFRAG,
                                              ln_g + G, ln_b + G, nullptr, out);
}
